// Glm4MoeMoE_55765855371548
// MI455X (gfx1250) — compile-verified, matrix operand order checked by reading, not run on hardware
//
#include <hip/hip_runtime.h>
#include <hip/hip_bf16.h>

// ---------------- problem constants (from reference) ----------------
#define E_ 32
#define K_ 4
#define H_ 2048
#define I_ 1024
#define T_ 2048          // B*S = 2*1024

typedef __attribute__((ext_vector_type(16))) __bf16 v16bf;
typedef __attribute__((ext_vector_type(8)))  float  v8f;
typedef unsigned int u32x4 __attribute__((ext_vector_type(4)));
typedef int          i32x8 __attribute__((ext_vector_type(8)));
typedef int          i32x4 __attribute__((ext_vector_type(4)));

union BF16x16 {
    v16bf v;
    unsigned short u[16];
    unsigned long long q[4];
};

__device__ __forceinline__ unsigned short f2bf(float f) {
    // round-to-nearest-even f32 -> bf16 (bit trick; NaN not special-cased)
    unsigned u = __float_as_uint(f);
    unsigned r = u + 0x7FFFu + ((u >> 16) & 1u);
    return (unsigned short)(r >> 16);
}

__device__ __forceinline__ float fast_sigmoid(float x) {
    return __builtin_amdgcn_rcpf(1.f + __expf(-x));    // v_rcp_f32, no div chain
}

__device__ __forceinline__ v8f wmma_bf16(const v16bf& a, const v16bf& b, v8f c) {
    return __builtin_amdgcn_wmma_f32_16x16x32_bf16(
        /*neg_a=*/false, a, /*neg_b=*/false, b,
        /*c_mod=*/(short)0, c, /*reuse_a=*/false, /*reuse_b=*/false);
}

// Issue one TDM load: tile (rows x cols) of f32 at gaddr (row pitch `stride` elems)
// into LDS at lds_off. padded!=0 -> +1 DWORD LDS padding per 256-DWORD row (pitch 257).
__device__ __forceinline__ void tdm_load_f32_tile(unsigned long long gaddr,
                                                  unsigned lds_off,
                                                  unsigned cols, unsigned rows,
                                                  unsigned stride, int padded) {
    // D# group 0: count=1 | lds_addr | global_addr[56:0] | type=2
    u32x4 g0 = { 1u,
                 lds_off,
                 (unsigned)(gaddr & 0xFFFFFFFFull),
                 (unsigned)((gaddr >> 32) & 0x01FFFFFFull) | 0x80000000u };
    // D# group 1
    unsigned d0 = (2u << 16);                                  // data_size = 4B
    if (padded) d0 |= (1u << 20) | (7u << 22);                 // pad: 1 DW per 256 DW
    i32x8 g1 = { (int)d0,
                 (int)(cols << 16),        // tensor_dim0[15:0]  (== tile_dim0)
                 (int)(rows << 16),        // tensor_dim0 hi=0 | tensor_dim1[15:0]
                 (int)(cols << 16),        // tensor_dim1 hi=0 | tile_dim0
                 (int)rows,                // tile_dim1 | tile_dim2=0
                 (int)stride,              // tensor_dim0_stride[31:0]
                 0, 0 };
    i32x4 gz4 = { 0, 0, 0, 0 };            // groups 2/3 unused (2D tile)
    i32x8 gz8 = { 0, 0, 0, 0, 0, 0, 0, 0 };
    __builtin_amdgcn_tensor_load_to_lds(g0, g1, gz4, gz4, gz8, 0);
}

__device__ __forceinline__ unsigned lds_byte_off(const void* p) {
    // LDS generic pointers: addr[31:0] is the LDS byte offset (LDS aperture)
    return (unsigned)(unsigned long long)p;
}

// ---------------- kernel 1: x f32 -> bf16 ----------------
__global__ __launch_bounds__(256) void cvt_kernel(const float* __restrict__ x,
                                                  unsigned short* __restrict__ xbf,
                                                  int n) {
    int i = (blockIdx.x * 256 + threadIdx.x) * 4;
    if (i + 3 < n) {
        float4 f = *(const float4*)(x + i);
        ushort4 p = make_ushort4(f2bf(f.x), f2bf(f.y), f2bf(f.z), f2bf(f.w));
        *(ushort4*)(xbf + i) = p;
    }
}

// ---------------- kernel 2: router (WMMA logits + top-4 + scatter) ----------------
__global__ __launch_bounds__(32) void router_kernel(
    const unsigned short* __restrict__ xbf,   // [T,H] bf16
    const float* __restrict__ rw,             // [E,H] f32
    const float* __restrict__ bias,           // [E]
    int* __restrict__ cnt,                    // [E] (zeroed)
    int* __restrict__ toks,                   // [E,T]
    float* __restrict__ wl)                   // [E,T]
{
    __shared__ float sc[16][32];
    const int lane = threadIdx.x;
    const int lmod = lane & 15, lhi = lane >> 4;
    const int t0 = blockIdx.x * 16;

    v8f c0 = {}, c1 = {};
    for (int k0 = 0; k0 < H_; k0 += 32) {
        BF16x16 a;
        {
            const unsigned long long* pa =
                (const unsigned long long*)(xbf + (size_t)(t0 + lmod) * H_ + k0 + lhi * 16);
            a.q[0] = pa[0]; a.q[1] = pa[1]; a.q[2] = pa[2]; a.q[3] = pa[3];
        }
        BF16x16 b0, b1;
        {
            const float4* p0 = (const float4*)(rw + (size_t)lmod * H_ + k0 + lhi * 16);
            const float4* p1 = (const float4*)(rw + (size_t)(16 + lmod) * H_ + k0 + lhi * 16);
#pragma unroll
            for (int j = 0; j < 4; ++j) {
                float4 f0 = p0[j], f1 = p1[j];
                b0.u[4 * j + 0] = f2bf(f0.x); b0.u[4 * j + 1] = f2bf(f0.y);
                b0.u[4 * j + 2] = f2bf(f0.z); b0.u[4 * j + 3] = f2bf(f0.w);
                b1.u[4 * j + 0] = f2bf(f1.x); b1.u[4 * j + 1] = f2bf(f1.y);
                b1.u[4 * j + 2] = f2bf(f1.z); b1.u[4 * j + 3] = f2bf(f1.w);
            }
        }
        c0 = wmma_bf16(a.v, b0.v, c0);
        c1 = wmma_bf16(a.v, b1.v, c1);
    }
#pragma unroll
    for (int r = 0; r < 8; ++r) {
        int m = r + 8 * lhi;
        sc[m][lmod]      = fast_sigmoid(c0[r]);
        sc[m][16 + lmod] = fast_sigmoid(c1[r]);
    }
    __syncthreads();

    if (lane < 16) {
        const int t = t0 + lane;
        float s[E_];
#pragma unroll
        for (int e = 0; e < E_; ++e) s[e] = sc[lane][e];
        unsigned mask = 0; int ch[K_]; float wv[K_]; float ssum = 0.f;
#pragma unroll
        for (int k = 0; k < K_; ++k) {         // selection uses scores+bias
            float best = -1e30f; int bi = 0;
            for (int e = 0; e < E_; ++e) {
                if (mask & (1u << e)) continue;
                float v = s[e] + bias[e];
                if (v > best) { best = v; bi = e; }
            }
            mask |= 1u << bi; ch[k] = bi; wv[k] = s[bi]; ssum += s[bi];
        }
        float inv = __builtin_amdgcn_rcpf(ssum + 1e-20f);   // RSF == 1.0
#pragma unroll
        for (int k = 0; k < K_; ++k) {
            int slot = atomicAdd(&cnt[ch[k]], 1);
            toks[ch[k] * T_ + slot] = t;
            wl[ch[k] * T_ + slot] = wv[k] * inv;
        }
    }
}

// ---------------- kernel 3: padded offsets (32-row aligned) ----------------
__global__ void offs_kernel(const int* __restrict__ cnt, int* __restrict__ offs) {
    if (threadIdx.x == 0 && blockIdx.x == 0) {
        int run = 0;
        for (int e = 0; e < E_; ++e) { offs[e] = run; run += ((cnt[e] + 31) >> 5) << 5; }
        offs[E_] = run;   // shared-expert act base / routed total
    }
}

// ---------------- kernel 4: gate/up GEMM + silu -> act (bf16) ----------------
// grid: (64 rowTiles of 32, I/256, E+1); block 256 thr = 8 waves
// W chunks staged via TDM (tensor_load_to_lds) as f32, converted lane-side.
__global__ __launch_bounds__(256) void gateup_kernel(
    const unsigned short* __restrict__ xbf,
    const float* __restrict__ wg, const float* __restrict__ wu,     // [E][H][I]
    const float* __restrict__ swg, const float* __restrict__ swu,   // [H][I]
    const int* __restrict__ cnt, const int* __restrict__ offs,
    const int* __restrict__ toks,
    unsigned short* __restrict__ act)                               // [rows][I] bf16
{
    const int e = blockIdx.z;
    const bool sh = (e == E_);
    const int n_e = sh ? T_ : cnt[e];
    const int m0 = blockIdx.x * 32;
    if (m0 >= n_e) return;
    const int nb0 = blockIdx.y * 256;
    const float* G = sh ? swg : (wg + (size_t)e * H_ * I_);
    const float* U = sh ? swu : (wu + (size_t)e * H_ * I_);
    const int base = sh ? offs[E_] : offs[e];

    __shared__ float lwg[32 * 256];          // 32 KB
    __shared__ float lwu[32 * 256];          // 32 KB

    const int tid = threadIdx.x;
    const int lane = tid & 31, w = tid >> 5;
    const int mt = w & 1, cg = w >> 1;
    const int lmod = lane & 15, lhi = lane >> 4;

    const int slot = m0 + mt * 16 + lmod;
    const int tok = sh ? slot : (slot < n_e ? toks[e * T_ + slot] : 0);
    const unsigned short* xrow = xbf + (size_t)tok * H_;

    const unsigned ldsG = lds_byte_off(lwg);
    const unsigned ldsU = lds_byte_off(lwu);

    v8f accg[4] = {}, accu[4] = {};
    for (int k0 = 0; k0 < H_; k0 += 32) {
        __syncthreads();                     // everyone done reading previous chunk
        if (w == 0) {                        // one TDM issue per chunk (wave 0)
            unsigned long long ga = (unsigned long long)(const void*)(G + (size_t)k0 * I_ + nb0);
            unsigned long long ua = (unsigned long long)(const void*)(U + (size_t)k0 * I_ + nb0);
            tdm_load_f32_tile(ga, ldsG, 256, 32, I_, /*padded=*/0);
            tdm_load_f32_tile(ua, ldsU, 256, 32, I_, /*padded=*/0);
            __builtin_amdgcn_s_wait_tensorcnt(0);
        }
        __syncthreads();                     // tile visible to all waves

        BF16x16 a;
        const unsigned long long* pa = (const unsigned long long*)(xrow + k0 + lhi * 16);
        a.q[0] = pa[0]; a.q[1] = pa[1]; a.q[2] = pa[2]; a.q[3] = pa[3];

#pragma unroll
        for (int ct = 0; ct < 4; ++ct) {
            const int n = cg * 64 + ct * 16 + lmod;
            BF16x16 bg, bu;
#pragma unroll
            for (int j = 0; j < 16; ++j) {
                bg.u[j] = f2bf(lwg[(lhi * 16 + j) * 256 + n]);
                bu.u[j] = f2bf(lwu[(lhi * 16 + j) * 256 + n]);
            }
            accg[ct] = wmma_bf16(a.v, bg.v, accg[ct]);
            accu[ct] = wmma_bf16(a.v, bu.v, accu[ct]);
        }
    }
    // epilogue: silu(gate) * up -> bf16 act, row-major (A-matrix-friendly)
#pragma unroll
    for (int ct = 0; ct < 4; ++ct) {
        const int col = nb0 + cg * 64 + ct * 16 + lmod;
#pragma unroll
        for (int r = 0; r < 8; ++r) {
            const int m = mt * 16 + r + 8 * lhi;
            float g = accg[ct][r], u = accu[ct][r];
            float h = g * fast_sigmoid(g) * u;          // silu(g) * u
            act[(size_t)(base + m0 + m) * I_ + col] = f2bf(h);
        }
    }
}

// ---------------- kernel 5: down GEMM, weighted scatter-add into out ----------------
// TDM staging with LDS row padding (pitch 257 DW) -> bank-conflict-free B reads.
__global__ __launch_bounds__(256) void down_kernel(
    const unsigned short* __restrict__ act,
    const float* __restrict__ wd, const float* __restrict__ swd,    // [E][I][H], [I][H]
    const int* __restrict__ cnt, const int* __restrict__ offs,
    const int* __restrict__ toks, const float* __restrict__ wl,
    float* __restrict__ out)                                        // [T,H] f32 (zeroed)
{
    const int e = blockIdx.z;
    const bool sh = (e == E_);
    const int n_e = sh ? T_ : cnt[e];
    const int m0 = blockIdx.x * 32;
    if (m0 >= n_e) return;
    const int nb0 = blockIdx.y * 256;
    const float* D = sh ? swd : (wd + (size_t)e * I_ * H_);
    const int base = sh ? offs[E_] : offs[e];

    __shared__ float lw[32 * 257 + 1];       // padded pitch 257 DW (TDM pad_enable)

    const int tid = threadIdx.x;
    const int lane = tid & 31, w = tid >> 5;
    const int mt = w & 1, cg = w >> 1;
    const int lmod = lane & 15, lhi = lane >> 4;

    const int slotA = m0 + mt * 16 + lmod;
    const unsigned short* arow = act + (size_t)(base + slotA) * I_;
    const unsigned ldsD = lds_byte_off(lw);

    v8f acc[4] = {};
    for (int k0 = 0; k0 < I_; k0 += 32) {
        __syncthreads();
        if (w == 0) {
            unsigned long long da = (unsigned long long)(const void*)(D + (size_t)k0 * H_ + nb0);
            tdm_load_f32_tile(da, ldsD, 256, 32, H_, /*padded=*/1);
            __builtin_amdgcn_s_wait_tensorcnt(0);
        }
        __syncthreads();

        BF16x16 a;
        const unsigned long long* pa = (const unsigned long long*)(arow + k0 + lhi * 16);
        a.q[0] = pa[0]; a.q[1] = pa[1]; a.q[2] = pa[2]; a.q[3] = pa[3];

#pragma unroll
        for (int ct = 0; ct < 4; ++ct) {
            const int n = cg * 64 + ct * 16 + lmod;
            BF16x16 b;
#pragma unroll
            for (int j = 0; j < 16; ++j) b.u[j] = f2bf(lw[(lhi * 16 + j) * 257 + n]);
            acc[ct] = wmma_bf16(a.v, b.v, acc[ct]);
        }
    }
    // epilogue: per-row combine weight, atomic scatter into out[token]
#pragma unroll
    for (int r = 0; r < 8; ++r) {
        const int m = mt * 16 + r + 8 * lhi;
        const int slot = m0 + m;
        float wrow; int tok;
        if (sh)               { wrow = 1.f;                tok = slot; }
        else if (slot < n_e)  { wrow = wl[e * T_ + slot];  tok = toks[e * T_ + slot]; }
        else                  { wrow = 0.f;                tok = 0; }
#pragma unroll
        for (int ct = 0; ct < 4; ++ct) {
            const int col = nb0 + cg * 64 + ct * 16 + lmod;
            atomicAdd(&out[(size_t)tok * H_ + col], acc[ct][r] * wrow);
        }
    }
}

// ---------------- host launcher ----------------
extern "C" void kernel_launch(void* const* d_in, const int* in_sizes, int n_in,
                              void* d_out, int out_size, void* d_ws, size_t ws_size,
                              hipStream_t stream) {
    const float* x    = (const float*)d_in[0];   // [B,S,H]
    const float* rw   = (const float*)d_in[1];   // [E,H]
    const float* bias = (const float*)d_in[2];   // [E]
    const float* wg   = (const float*)d_in[3];   // [E,H,I]
    const float* wu   = (const float*)d_in[4];   // [E,H,I]
    const float* wdn  = (const float*)d_in[5];   // [E,I,H]
    const float* swg  = (const float*)d_in[6];   // [H,I]
    const float* swu  = (const float*)d_in[7];   // [H,I]
    const float* swd  = (const float*)d_in[8];   // [I,H]
    float* out = (float*)d_out;

    // workspace layout (bytes)
    char* ws = (char*)d_ws;
    unsigned short* xbf = (unsigned short*)(ws);              //  8,388,608  x bf16
    int*   cnt  = (int*)  (ws + 8388608);                     //        256  counters
    int*   offs = (int*)  (ws + 8388864);                     //        256  padded offsets
    int*   toks = (int*)  (ws + 8389120);                     //    262,144  [E][T]
    float* wl   = (float*)(ws + 8651264);                     //    262,144  [E][T]
    unsigned short* act = (unsigned short*)(ws + 8913408);    // 23,068,672  act bf16
    // total ~32 MB

    (void)hipMemsetAsync(cnt, 0, 256, stream);
    (void)hipMemsetAsync(d_out, 0, (size_t)T_ * H_ * sizeof(float), stream);

    cvt_kernel<<<(T_ * H_) / (256 * 4), 256, 0, stream>>>(x, xbf, T_ * H_);
    router_kernel<<<T_ / 16, 32, 0, stream>>>(xbf, rw, bias, cnt, toks, wl);
    offs_kernel<<<1, 32, 0, stream>>>(cnt, offs);
    gateup_kernel<<<dim3(64, I_ / 256, E_ + 1), 256, 0, stream>>>(
        xbf, wg, wu, swg, swu, cnt, offs, toks, act);
    down_kernel<<<dim3(64, H_ / 256, E_ + 1), 256, 0, stream>>>(
        act, wdn, swd, cnt, offs, toks, wl, out);
}
